// CharEmbeddingCNN_82403242541716
// MI455X (gfx1250) — matrix operand order repaired and checked by reading, not compiled, not run on hardware
//
#include <hip/hip_runtime.h>

typedef _Float16 h8  __attribute__((ext_vector_type(8)));
typedef _Float16 h16 __attribute__((ext_vector_type(16)));
typedef float    f8  __attribute__((ext_vector_type(8)));

#define NWORDS 8192
#define SEQ_L 20
#define DIM 256
#define WPG 16                          // words per workgroup (WMMA N-tile)
#define ROW_B (DIM * 2)                 // 512 bytes per t-row (f16)
#define WORD_STRIDE_B (SEQ_L * ROW_B)   // 10240 bytes per word in LDS

__device__ __forceinline__ h16 cat8(h8 lo, h8 hi) {
  return __builtin_shufflevector(lo, hi, 0,1,2,3,4,5,6,7,8,9,10,11,12,13,14,15);
}

// One conv kernel size K: for each valid position t, GEMM over K-dim = 256*K
// (c in steps of 32, taps j), running max over t, then bias + merge into result.
template<int K>
__device__ __forceinline__ void conv_max(
    const _Float16* __restrict__ W,     // f16 weights, layout [j][d:256][c:256]
    const float* __restrict__ bias,
    const unsigned char* xeLds, int nCol, int half, int dbase,
    f8& result0, f8& result1)
{
  f8 m0, m1;
  #pragma unroll
  for (int i = 0; i < 8; ++i) { m0[i] = -3.0e38f; m1[i] = -3.0e38f; }

  for (int t = 0; t <= SEQ_L - K; ++t) {
    f8 acc0 = {};
    f8 acc1 = {};
    #pragma unroll
    for (int j = 0; j < K; ++j) {
      const unsigned char* brow = xeLds + nCol * WORD_STRIDE_B + (t + j) * ROW_B;
      const _Float16* wrow0 = W + ((j * 256) + dbase + nCol) * 256;  // A row m = lane&15
      const _Float16* wrow1 = wrow0 + 16 * 256;                      // second d-tile
      const int sw = nCol << 4;                                      // LDS bank swizzle
      #pragma unroll
      for (int cs = 0; cs < 8; ++cs) {
        const int c0 = cs * 32;
        // B fragment (32x16 K-major): per lane K groups {h*8..h*8+7} and {16+h*8..23+h*8}
        const int bb = c0 * 2 + half * 16;
        h8 blo = *(const h8*)(brow + ( bb        ^ sw));
        h8 bhi = *(const h8*)(brow + ((bb + 32)  ^ sw));
        h16 B = cat8(blo, bhi);
        // A fragments (16x32): per lane K groups {h*8..h*8+7} and {16+h*8..23+h*8}
        h8 a0lo = *(const h8*)(wrow0 + c0 + half * 8);
        h8 a0hi = *(const h8*)(wrow0 + c0 + 16 + half * 8);
        h16 A0 = cat8(a0lo, a0hi);
        h8 a1lo = *(const h8*)(wrow1 + c0 + half * 8);
        h8 a1hi = *(const h8*)(wrow1 + c0 + 16 + half * 8);
        h16 A1 = cat8(a1lo, a1hi);
        acc0 = __builtin_amdgcn_wmma_f32_16x16x32_f16(false, A0, false, B,
                                                      (short)0, acc0, false, false);
        acc1 = __builtin_amdgcn_wmma_f32_16x16x32_f16(false, A1, false, B,
                                                      (short)0, acc1, false, false);
      }
    }
    #pragma unroll
    for (int i = 0; i < 8; ++i) {
      m0[i] = fmaxf(m0[i], acc0[i]);
      m1[i] = fmaxf(m1[i], acc1[i]);
    }
  }
  // bias is position-invariant: add after the max, merge across convs
  #pragma unroll
  for (int r = 0; r < 8; ++r) {
    const int d0 = dbase + half * 8 + r;          // C tile: M = r + 8*half
    result0[r] = fmaxf(result0[r], m0[r] + bias[d0]);
    result1[r] = fmaxf(result1[r], m1[r] + bias[d0 + 16]);
  }
}

// Convert emb -> f16 (same layout) and repack conv weights w[d][c][j] (f32)
// -> [j][d][c] (f16) so A-frags are contiguous 16B loads.
// 65536 + 786432 = 851968 elements total.
__global__ __launch_bounds__(256) void charcnn_prep(
    const float* __restrict__ emb,
    const float* __restrict__ w3, const float* __restrict__ w4,
    const float* __restrict__ w5,
    _Float16* __restrict__ embh,
    _Float16* __restrict__ w3h, _Float16* __restrict__ w4h,
    _Float16* __restrict__ w5h)
{
  const int i = blockIdx.x * 256 + threadIdx.x;
  if (i < 65536) {
    embh[i] = (_Float16)emb[i];
    return;
  }
  const int iw = i - 65536;
  int k, e;
  const float* src;
  _Float16* dst;
  if (iw < 3 * 65536)      { k = 3; src = w3; dst = w3h; e = iw; }
  else if (iw < 7 * 65536) { k = 4; src = w4; dst = w4h; e = iw - 3 * 65536; }
  else                     { k = 5; src = w5; dst = w5h; e = iw - 7 * 65536; }
  const int j = e >> 16;
  const int rem = e & 65535;
  const int d = rem >> 8;
  const int c = rem & 255;
  dst[e] = (_Float16)src[(d * 256 + c) * k + j];
}

__global__ __launch_bounds__(256) void charcnn_main(
    const int* __restrict__ x, const int* __restrict__ lens,
    const _Float16* __restrict__ embh,
    const _Float16* __restrict__ w3h, const _Float16* __restrict__ w4h,
    const _Float16* __restrict__ w5h,
    const float* __restrict__ b3, const float* __restrict__ b4,
    const float* __restrict__ b5,
    float* __restrict__ out)
{
  __shared__ __align__(16) unsigned char xeLds[WPG * WORD_STRIDE_B]; // 160 KB

  const int tid = threadIdx.x;
  const int lane = tid & 31;
  const int wave = tid >> 5;
  const int wordBase = blockIdx.x * WPG;

  // ---- Phase 0: gather f16 embedding rows into LDS [word][t][c], swizzled ----
  for (int r = wave; r < WPG * SEQ_L; r += 8) {
    const int word = r / SEQ_L;
    const int t = r - word * SEQ_L;
    const int ch = x[(wordBase + word) * SEQ_L + t];
    h8 h = *(const h8*)(embh + ch * DIM + lane * 8);      // 16B coalesced
    const int chunk = (lane * 16) ^ (word << 4);          // per-16B-chunk XOR swizzle
    *(h8*)(xeLds + word * WORD_STRIDE_B + t * ROW_B + chunk) = h;
  }
  __syncthreads();

  // ---- Phase 1: WMMA conv + max-pool; wave w owns d in [32w, 32w+32) ----
  const int half = lane >> 4;
  const int nCol = lane & 15;
  const int dbase = wave * 32;

  f8 result0 = {};   // init 0 == built-in ReLU
  f8 result1 = {};
  conv_max<3>(w3h, b3, xeLds, nCol, half, dbase, result0, result1);
  conv_max<4>(w4h, b4, xeLds, nCol, half, dbase, result0, result1);
  conv_max<5>(w5h, b5, xeLds, nCol, half, dbase, result0, result1);

  // ---- Store: C-tile element (m = r + 8*half, n = lane&15) ----
  const int gword = wordBase + nCol;
  const bool valid = lens[gword] != 0;
  float* orow = out + gword * DIM + dbase + half * 8;
  #pragma unroll
  for (int r = 0; r < 8; ++r) {
    orow[r]      = valid ? result0[r] : 0.0f;
    orow[16 + r] = valid ? result1[r] : 0.0f;
  }
}

extern "C" void kernel_launch(void* const* d_in, const int* in_sizes, int n_in,
                              void* d_out, int out_size, void* d_ws, size_t ws_size,
                              hipStream_t stream) {
  const int*   x    = (const int*)d_in[0];
  const int*   lens = (const int*)d_in[1];
  const float* emb  = (const float*)d_in[2];
  const float* w3   = (const float*)d_in[3];
  const float* b3   = (const float*)d_in[4];
  const float* w4   = (const float*)d_in[5];
  const float* b4   = (const float*)d_in[6];
  const float* w5   = (const float*)d_in[7];
  const float* b5   = (const float*)d_in[8];
  float* out = (float*)d_out;

  _Float16* embh = (_Float16*)d_ws;          // 65536 f16  (128 KB)
  _Float16* w3h  = embh + 65536;             // 3*65536 f16
  _Float16* w4h  = w3h + 3 * 65536;          // 4*65536 f16
  _Float16* w5h  = w4h + 4 * 65536;          // 5*65536 f16  (total ~1.7 MB)

  charcnn_prep<<<(13 * 65536) / 256, 256, 0, stream>>>(emb, w3, w4, w5,
                                                       embh, w3h, w4h, w5h);
  charcnn_main<<<NWORDS / WPG, 256, 0, stream>>>(x, lens, embh, w3h, w4h, w5h,
                                                 b3, b4, b5, out);
}
